// GroupedQueryAttention_35373350650261
// MI455X (gfx1250) — matrix operand order repaired and checked by reading, not compiled, not run on hardware
//
#include <hip/hip_runtime.h>
#include <stdint.h>

// ---------------- types ----------------
typedef __attribute__((ext_vector_type(16))) __bf16 bf16x16;
typedef __attribute__((ext_vector_type(8)))  __bf16 bf16x8;
typedef __attribute__((ext_vector_type(4)))  __bf16 bf16x4;
typedef __attribute__((ext_vector_type(8)))  float  f32x8;
typedef __attribute__((ext_vector_type(4)))  float  f32x4;

union BF16Frag {
    bf16x16 v;
    bf16x8  h[2];
    __bf16  e[16];
};

__device__ __forceinline__ __bf16 f2bf(float f) {
    uint32_t u = __builtin_bit_cast(uint32_t, f);
    u += 0x7FFFu + ((u >> 16) & 1u);                // round-to-nearest-even
    uint16_t s = (uint16_t)(u >> 16);
    return __builtin_bit_cast(__bf16, s);
}

__device__ __forceinline__ f32x8 wmma_bf16f32(const BF16Frag& a, const BF16Frag& b, f32x8 c) {
    // D = A(16x32 bf16) x B(32x16 bf16) + C(16x16 f32)
    return __builtin_amdgcn_wmma_f32_16x16x32_bf16(false, a.v, false, b.v, (short)0, c, false, false);
}

// ---------------- f32 -> bf16 pack (row-major copy) ----------------
__global__ __launch_bounds__(256)
void cvt_f32_bf16(const float* __restrict__ src, __bf16* __restrict__ dst, int n4) {
    int i = blockIdx.x * 256 + threadIdx.x;
    if (i >= n4) return;
    f32x4 v = *(const f32x4*)(src + 4 * (size_t)i);
    bf16x4 o;
    o[0] = f2bf(v[0]); o[1] = f2bf(v[1]); o[2] = f2bf(v[2]); o[3] = f2bf(v[3]);
    *(bf16x4*)(dst + 4 * (size_t)i) = o;
}

// ---------------- f32 [R][C] -> bf16 [C][R] transpose (weights, one-time) ----------------
__global__ __launch_bounds__(256)
void transpose_f32_bf16(const float* __restrict__ src, __bf16* __restrict__ dst, int R, int C) {
    __shared__ float tile[32][33];
    const int c0 = blockIdx.x * 32, r0 = blockIdx.y * 32;
    const int tx = threadIdx.x & 31, ty = threadIdx.x >> 5;   // 32 x 8
    #pragma unroll
    for (int i = 0; i < 32; i += 8)
        tile[ty + i][tx] = src[(size_t)(r0 + ty + i) * C + c0 + tx];
    __syncthreads();
    #pragma unroll
    for (int i = 0; i < 32; i += 8)
        dst[(size_t)(c0 + ty + i) * R + r0 + tx] = f2bf(tile[tx][ty + i]);
}

// ---------------- bf16 GEMM: C[M,N] = A[M,K] * BT[N,K]^T + bias[N] ----------------
// 256 threads (8 waves), C tile 128x128, wave tile 64x32 (4x2 WMMA tiles), K-step 32.
// Both tiles staged with GLOBAL_LOAD_ASYNC_TO_LDS_B128, 2-deep pipeline, 1 barrier/iter.
template<bool OUT_BF16>
__global__ __launch_bounds__(256)
void gemm_bf16(const __bf16* __restrict__ A,   // [M][K] row-major
               const __bf16* __restrict__ BT,  // [N][K] row-major (B pre-transposed)
               const float* __restrict__ bias,
               void* __restrict__ Cout, int M, int N, int K)
{
    (void)M;
    constexpr int LD = 40;                       // padded K-stride (80B, 16B aligned)
    constexpr int TILE = 128 * LD;               // elems per buffer
    __shared__ alignas(16) __bf16 As[2 * TILE];  // [m][k], double buffered
    __shared__ alignas(16) __bf16 Bs[2 * TILE];  // [n][k], double buffered

    const int t = threadIdx.x;
    const int lane = t & 31, wave = t >> 5;
    const int l15 = lane & 15, hi = lane >> 4;
    const int blockM = blockIdx.y * 128, blockN = blockIdx.x * 128;
    const int wm = (wave & 1) * 64, wn = (wave >> 1) * 32;

    f32x8 acc[4][2];
    #pragma unroll
    for (int mi = 0; mi < 4; ++mi)
        #pragma unroll
        for (int ni = 0; ni < 2; ++ni)
            #pragma unroll
            for (int r = 0; r < 8; ++r) acc[mi][ni][r] = 0.0f;

    // staging map: 128 rows x 32 k per tile; thread -> (row, 16-elem chunk)
    const int srow = t >> 1, skoff = (t & 1) * 16;
    const __bf16* gA = A  + (size_t)(blockM + srow) * K + skoff;
    const __bf16* gB = BT + (size_t)(blockN + srow) * K + skoff;
    const uint32_t laA = (uint32_t)(uintptr_t)(As + srow * LD + skoff);  // LDS byte addr
    const uint32_t laB = (uint32_t)(uintptr_t)(Bs + srow * LD + skoff);

    auto issue = [&](int k0, int buf) {
        uint64_t ga = (uint64_t)(uintptr_t)(gA + k0);
        uint64_t gb = (uint64_t)(uintptr_t)(gB + k0);
        uint32_t la = laA + (uint32_t)buf * (TILE * 2);
        uint32_t lb = laB + (uint32_t)buf * (TILE * 2);
        // INST_OFFSET applies to BOTH lds and global address (ISA 15.18.3)
        asm volatile(
            "global_load_async_to_lds_b128 %0, %2, off\n\t"
            "global_load_async_to_lds_b128 %0, %2, off offset:16\n\t"
            "global_load_async_to_lds_b128 %1, %3, off\n\t"
            "global_load_async_to_lds_b128 %1, %3, off offset:16"
            :: "v"(la), "v"(lb), "v"(ga), "v"(gb) : "memory");
    };

    const int niter = K >> 5;
    issue(0, 0);
    for (int i = 0; i < niter; ++i) {
        asm volatile("s_wait_asynccnt 0x0" ::: "memory");  // tile i landed (this wave)
        __syncthreads();                                    // all waves landed; prev reads done
        if (i + 1 < niter) issue((i + 1) * 32, (i + 1) & 1);// prefetch next while computing

        const __bf16* as = As + (i & 1) * TILE;
        const __bf16* bs = Bs + (i & 1) * TILE;
        BF16Frag af[4], bfr[2];
        #pragma unroll
        for (int mi = 0; mi < 4; ++mi) {   // A frag: row=l15, K = e + 16*hi (contiguous)
            const __bf16* p = as + (wm + mi * 16 + l15) * LD;
            af[mi].h[0] = *(const bf16x8*)(p + 16 * hi);
            af[mi].h[1] = *(const bf16x8*)(p + 16 * hi + 8);
        }
        #pragma unroll
        for (int ni = 0; ni < 2; ++ni) {   // B frag: col=l15, K = e + 16*hi (contiguous)
            const __bf16* p = bs + (wn + ni * 16 + l15) * LD + 16 * hi;
            bfr[ni].h[0] = *(const bf16x8*)p;
            bfr[ni].h[1] = *(const bf16x8*)(p + 8);
        }
        #pragma unroll
        for (int mi = 0; mi < 4; ++mi)
            #pragma unroll
            for (int ni = 0; ni < 2; ++ni)
                acc[mi][ni] = wmma_bf16f32(af[mi], bfr[ni], acc[mi][ni]);
    }

    // epilogue: C layout row = r + 8*hi, col = l15
    #pragma unroll
    for (int ni = 0; ni < 2; ++ni) {
        const int col = blockN + wn + ni * 16 + l15;
        const float bv = bias[col];
        #pragma unroll
        for (int mi = 0; mi < 4; ++mi) {
            #pragma unroll
            for (int r = 0; r < 8; ++r) {
                const int row = blockM + wm + mi * 16 + r + 8 * hi;
                const float val = acc[mi][ni][r] + bv;
                if (OUT_BF16) ((__bf16*)Cout)[(size_t)row * N + col] = f2bf(val);
                else          ((float*)Cout)[(size_t)row * N + col] = val;
            }
        }
    }
}

// ---------------- flash attention (causal, V == K) ----------------
// grid: (S/64, NUM_HEADS); 128 threads (4 waves). Wave owns 16 query rows.
// Key blocks of 32; online softmax; O accumulated in registers (16 dim-tiles x v8f).
#define HD   256
#define QSTR 2048

__global__ __launch_bounds__(128)
void flash_gqa(const __bf16* __restrict__ Qb,  // [S][2048] (head h at cols h*256..)
               const __bf16* __restrict__ Kb,  // [S][256]
               __bf16* __restrict__ Ob,        // [S][2048]
               int S)
{
    (void)S;
    __shared__ alignas(16) __bf16 Ks[32 * 264];   // K tile row-major [key][dim], padded
    __shared__ alignas(16) __bf16 Kst[256 * 40];  // K tile transposed [dim][key], padded
    __shared__ alignas(16) __bf16 Pl[4 * 16 * 40];// per-wave P staging [row][key]

    const int t = threadIdx.x;
    const int lane = t & 31, wave = t >> 5;
    const int l15 = lane & 15, hi = lane >> 4;
    const int head = blockIdx.y;
    const int qb = blockIdx.x;
    const int qrow0 = qb * 64 + wave * 16;

    // preload Q fragments: 8 dim k-tiles of 16x32
    BF16Frag qf[8];
    {
        const __bf16* qp = Qb + (size_t)(qrow0 + l15) * QSTR + head * HD;
        #pragma unroll
        for (int kt = 0; kt < 8; ++kt) {
            qf[kt].h[0] = *(const bf16x8*)(qp + 32 * kt + 16 * hi);
            qf[kt].h[1] = *(const bf16x8*)(qp + 32 * kt + 16 * hi + 8);
        }
    }

    f32x8 o[16];
    #pragma unroll
    for (int nt = 0; nt < 16; ++nt)
        #pragma unroll
        for (int r = 0; r < 8; ++r) o[nt][r] = 0.0f;

    float mrow[8], lrow[8];
    #pragma unroll
    for (int r = 0; r < 8; ++r) { mrow[r] = -3.0e38f; lrow[r] = 0.0f; }

    const float scale = 0.0625f;                  // 1/sqrt(256)
    const float log2e = 1.44269504088896f;

    const int kbmax = 2 * qb + 1;                 // causal: keys <= qb*64+63
    const int krow = t >> 2, koff = (t & 3) * 64; // cooperative K-tile load mapping

    for (int kb = 0; kb <= kbmax; ++kb) {
        __syncthreads();
        {   // stage K tile: rows kb*32..+31 -> Ks (row major) and Kst (transposed)
            const __bf16* g = Kb + (size_t)(kb * 32 + krow) * HD + koff;
            #pragma unroll
            for (int i = 0; i < 8; ++i) {
                BF16Frag c;
                c.h[0] = *(const bf16x8*)(g + i * 8);
                *(bf16x8*)(Ks + krow * 264 + koff + i * 8) = c.h[0];
                #pragma unroll
                for (int j = 0; j < 8; ++j)
                    Kst[(koff + i * 8 + j) * 40 + krow] = c.e[j];
            }
        }
        __syncthreads();

        // S = Q * K^T : 2 key tiles, 8 dim k-steps
        f32x8 s[2];
        #pragma unroll
        for (int kt = 0; kt < 2; ++kt)
            #pragma unroll
            for (int r = 0; r < 8; ++r) s[kt][r] = 0.0f;

        #pragma unroll
        for (int kd = 0; kd < 8; ++kd) {
            #pragma unroll
            for (int kt = 0; kt < 2; ++kt) {
                BF16Frag b;
                const __bf16* p = Ks + (kt * 16 + l15) * 264 + 32 * kd + 16 * hi;
                b.h[0] = *(const bf16x8*)p;
                b.h[1] = *(const bf16x8*)(p + 8);
                s[kt] = wmma_bf16f32(qf[kd], b, s[kt]);
            }
        }

        // online softmax over rows (row = r + 8*hi); row spans 16 lanes per half
        const bool maymask = (kb * 32 + 31) > qrow0;
        #pragma unroll
        for (int r = 0; r < 8; ++r) {
            const int grow = qrow0 + r + 8 * hi;
            float mx = -3.0e38f;
            #pragma unroll
            for (int kt = 0; kt < 2; ++kt) {
                float v = s[kt][r] * scale;
                if (maymask && (kb * 32 + kt * 16 + l15) > grow) v = -3.0e38f;
                s[kt][r] = v;
                mx = fmaxf(mx, v);
            }
            mx = fmaxf(mx, __shfl_xor(mx, 1));
            mx = fmaxf(mx, __shfl_xor(mx, 2));
            mx = fmaxf(mx, __shfl_xor(mx, 4));
            mx = fmaxf(mx, __shfl_xor(mx, 8));
            const float mnew  = fmaxf(mrow[r], mx);
            const float alpha = exp2f((mrow[r] - mnew) * log2e);
            float rs = 0.0f;
            #pragma unroll
            for (int kt = 0; kt < 2; ++kt) {
                float p = exp2f((s[kt][r] - mnew) * log2e);
                s[kt][r] = p;
                rs += p;
            }
            rs += __shfl_xor(rs, 1);
            rs += __shfl_xor(rs, 2);
            rs += __shfl_xor(rs, 4);
            rs += __shfl_xor(rs, 8);
            lrow[r] = lrow[r] * alpha + rs;
            mrow[r] = mnew;
            #pragma unroll
            for (int nt = 0; nt < 16; ++nt) o[nt][r] *= alpha;
        }

        // P (C-layout) -> LDS -> A-layout re-swizzle
        {
            __bf16* pw = Pl + wave * 16 * 40;
            #pragma unroll
            for (int r = 0; r < 8; ++r)
                #pragma unroll
                for (int kt = 0; kt < 2; ++kt)
                    pw[(r + 8 * hi) * 40 + kt * 16 + l15] = f2bf(s[kt][r]);
        }
        __syncthreads();

        // O += P * V (V = K tile; B operand from Kst, keys contiguous per dim)
        {
            BF16Frag a;
            const __bf16* pp = Pl + wave * 16 * 40 + l15 * 40;
            a.h[0] = *(const bf16x8*)(pp + 16 * hi);
            a.h[1] = *(const bf16x8*)(pp + 16 * hi + 8);
            #pragma unroll
            for (int nt = 0; nt < 16; ++nt) {
                BF16Frag b;
                const __bf16* vp = Kst + (nt * 16 + l15) * 40 + 16 * hi;
                b.h[0] = *(const bf16x8*)vp;
                b.h[1] = *(const bf16x8*)(vp + 8);
                o[nt] = wmma_bf16f32(a, b, o[nt]);
            }
        }
    }

    // normalize and write O (bf16, feeds the output projection)
    float inv[8];
    #pragma unroll
    for (int r = 0; r < 8; ++r) inv[r] = 1.0f / lrow[r];
    #pragma unroll
    for (int nt = 0; nt < 16; ++nt) {
        #pragma unroll
        for (int r = 0; r < 8; ++r) {
            const int row = qrow0 + r + 8 * hi;
            const int col = head * HD + nt * 16 + l15;
            Ob[(size_t)row * QSTR + col] = f2bf(o[nt][r] * inv[r]);
        }
    }
}

// ---------------- launch ----------------
extern "C" void kernel_launch(void* const* d_in, const int* in_sizes, int n_in,
                              void* d_out, int out_size, void* d_ws, size_t ws_size,
                              hipStream_t stream)
{
    (void)in_sizes; (void)n_in; (void)out_size; (void)ws_size;
    const float* x   = (const float*)d_in[0];
    const float* wq  = (const float*)d_in[1];
    const float* bq  = (const float*)d_in[2];
    const float* wkv = (const float*)d_in[3];
    const float* bkv = (const float*)d_in[4];
    const float* wo  = (const float*)d_in[5];
    const float* bo  = (const float*)d_in[6];
    // d_in[7] = mask: causal tril, implemented analytically in flash_gqa

    const int S = 4096, HID = 2048, NH = 8, D = 256;
    __bf16* ws    = (__bf16*)d_ws;
    __bf16* xb    = ws;                          // [4096][2048]
    __bf16* wqT   = xb   + (size_t)S * HID;      // [2048][2048]  (= wq^T, [N][K])
    __bf16* wkvT  = wqT  + (size_t)HID * NH * D; // [256][2048]   (= wkv^T)
    __bf16* woT   = wkvT + (size_t)HID * D;      // [2048][2048]  (= wo^T)
    __bf16* Qb    = woT  + (size_t)NH * D * HID; // [4096][2048]
    __bf16* Kbuf  = Qb   + (size_t)S * NH * D;   // [4096][256]
    __bf16* Ob    = Kbuf + (size_t)S * D;        // [4096][2048]
    // total ~67 MB of workspace

    {   // x: row-major bf16 copy (A operand stays [M][K])
        size_t n = (size_t)S * HID;
        cvt_f32_bf16<<<dim3((unsigned)(n / 4 / 256)), 256, 0, stream>>>(x, xb, (int)(n / 4));
    }
    // weights: fused f32->bf16 + transpose so GEMM B tiles are async-loadable row slabs
    transpose_f32_bf16<<<dim3((NH * D) / 32, HID / 32), 256, 0, stream>>>(wq,  wqT,  HID, NH * D);
    transpose_f32_bf16<<<dim3(D / 32,        HID / 32), 256, 0, stream>>>(wkv, wkvT, HID, D);
    transpose_f32_bf16<<<dim3(HID / 32, (NH * D) / 32), 256, 0, stream>>>(wo,  woT,  NH * D, HID);

    // Q = x*wq + bq  (bf16 out), K=V = x*wkv + bkv (bf16 out)
    gemm_bf16<true><<<dim3(2048 / 128, 4096 / 128), 256, 0, stream>>>(xb, wqT, bq, (void*)Qb, S, NH * D, HID);
    gemm_bf16<true><<<dim3(256 / 128, 4096 / 128), 256, 0, stream>>>(xb, wkvT, bkv, (void*)Kbuf, S, D, HID);

    // causal flash attention, GQA (1 KV head broadcast to 8 Q heads)
    flash_gqa<<<dim3(S / 64, NH), 128, 0, stream>>>(Qb, Kbuf, Ob, S);

    // out = O*wo + bo (f32 out)
    gemm_bf16<false><<<dim3(2048 / 128, 4096 / 128), 256, 0, stream>>>(Ob, woT, bo, d_out, S, HID, NH * D);
}
